// VanillaGNNLayer_64905545777776
// MI455X (gfx1250) — compile-verified, matrix operand order checked by reading, not run on hardware
//
#include <hip/hip_runtime.h>
#include <hip/hip_bf16.h>

// ---------------------------------------------------------------------------
// VanillaGNNLayer for MI455X (gfx1250, wave32, WMMA)
//   Phase 1: h = x @ W          via v_wmma_f32_16x16x32_f16 (f32 accumulate)
//            one wave computes a 16x64 slab: A loaded once, 4 independent
//            WMMAs per K-step (no RAW hazard between them), x traffic = 1x.
//   Phase 2: out = 0
//   Phase 3: out[r] += a_e * h[c_e]  via global_atomic_add_f32 (L2-resident)
// ---------------------------------------------------------------------------

#define DIM_IN  128
#define DIM_OUT 64

typedef __attribute__((ext_vector_type(16))) _Float16 v16h;
typedef __attribute__((ext_vector_type(8)))  float    v8f;

// ---------------------------------------------------------------------------
// Phase 1: dense GEMM. One wave32 computes one 16x64 slab of h.
// Block = 256 threads = 8 waves. K=128 -> 4 K-steps x 4 WMMAs.
// ---------------------------------------------------------------------------
__global__ __launch_bounds__(256) void gnn_gemm_kernel(
    const float* __restrict__ x,   // [n_nodes, 128] row-major
    const float* __restrict__ W,   // [128, 64]      row-major
    float* __restrict__ h,         // [n_nodes, 64]  row-major (workspace)
    int n_nodes)
{
    // Stage W transposed into LDS as f16: ldsWt[n*128 + k] = W[k*64 + n].
    // 64*128*2B = 16 KB (WGP has 320 KB). Each B-fragment load is two
    // contiguous 16B chunks -> ds_load_b128.
    __shared__ _Float16 ldsWt[DIM_OUT * DIM_IN];
    const int tid = threadIdx.x;
    for (int i = tid; i < DIM_IN * DIM_OUT; i += 256) {
        const int k = i >> 6;      // 0..127
        const int n = i & 63;      // 0..63
        ldsWt[n * DIM_IN + k] = (_Float16)W[i];
    }
    __syncthreads();

    const int mtiles = (n_nodes + 15) >> 4;          // 3125 for N=50000
    const int mtile  = blockIdx.x * 8 + (tid >> 5);  // one 16-row slab per wave
    if (mtile >= mtiles) return;

    const int lane   = tid & 31;
    const int laneHi = lane >> 4;                    // 0: lanes 0-15, 1: 16-31
    const int lane15 = lane & 15;

    int m = mtile * 16 + lane15;
    if (m >= n_nodes) m = n_nodes - 1;               // clamp loads (stores guarded)

    const float* ap = x + (size_t)m * DIM_IN;

    v8f acc0 = {}, acc1 = {}, acc2 = {}, acc3 = {};

    #pragma unroll
    for (int kt = 0; kt < 4; ++kt) {
        // ISA 7.12.2 16-bit A 16x32 layout:
        //   lanes 0-15 hold K-groups {0..7, 16..23}; lanes 16-31 {8..15, 24..31}
        const int ka = kt * 32 + laneHi * 8;         // first 8-K group
        const int kb = ka + 16;                      // second 8-K group

        union { v16h v; _Float16 e[16]; } A;
        #pragma unroll
        for (int j = 0; j < 8; ++j) {
            A.e[j]     = (_Float16)ap[ka + j];       // global_load_b128 x2 + cvt
            A.e[8 + j] = (_Float16)ap[kb + j];
        }

        // 4 n-tiles share the same A fragment: 4 independent WMMAs, no RAW.
        union { v16h v; _Float16 e[16]; } B0, B1, B2, B3;
        const _Float16* bp0 = ldsWt + (size_t)(0 * 16 + lane15) * DIM_IN;
        const _Float16* bp1 = ldsWt + (size_t)(1 * 16 + lane15) * DIM_IN;
        const _Float16* bp2 = ldsWt + (size_t)(2 * 16 + lane15) * DIM_IN;
        const _Float16* bp3 = ldsWt + (size_t)(3 * 16 + lane15) * DIM_IN;
        #pragma unroll
        for (int j = 0; j < 8; ++j) {
            B0.e[j] = bp0[ka + j];  B0.e[8 + j] = bp0[kb + j];   // ds_load_b128 x2
            B1.e[j] = bp1[ka + j];  B1.e[8 + j] = bp1[kb + j];
            B2.e[j] = bp2[ka + j];  B2.e[8 + j] = bp2[kb + j];
            B3.e[j] = bp3[ka + j];  B3.e[8 + j] = bp3[kb + j];
        }

        acc0 = __builtin_amdgcn_wmma_f32_16x16x32_f16(false, A.v, false, B0.v,
                                                      (short)0, acc0, false, false);
        acc1 = __builtin_amdgcn_wmma_f32_16x16x32_f16(false, A.v, false, B1.v,
                                                      (short)0, acc1, false, false);
        acc2 = __builtin_amdgcn_wmma_f32_16x16x32_f16(false, A.v, false, B2.v,
                                                      (short)0, acc2, false, false);
        acc3 = __builtin_amdgcn_wmma_f32_16x16x32_f16(false, A.v, false, B3.v,
                                                      (short)0, acc3, false, false);
    }

    // C/D layout: VGPR v -> row v (lanes 0-15) or row 8+v (lanes 16-31)
    const int row0 = mtile * 16 + laneHi * 8;
    #pragma unroll
    for (int v = 0; v < 8; ++v) {
        const int row = row0 + v;
        if (row < n_nodes) {
            float* hp = h + (size_t)row * DIM_OUT + lane15;
            hp[0]  = acc0[v];
            hp[16] = acc1[v];
            hp[32] = acc2[v];
            hp[48] = acc3[v];
        }
    }
}

// ---------------------------------------------------------------------------
// Phase 2: zero the accumulation target (harness poisons d_out with 0xAA).
// ---------------------------------------------------------------------------
__global__ __launch_bounds__(256) void gnn_zero_kernel(float4* __restrict__ out, int n4)
{
    const int i = blockIdx.x * 256 + threadIdx.x;
    if (i < n4) out[i] = make_float4(0.f, 0.f, 0.f, 0.f);
}

// ---------------------------------------------------------------------------
// Phase 3: COO scatter. One wave32 per edge; lane owns 2 of the 64 dims.
// Edge index forced into an SGPR so rows/cols/vals become scalar loads.
// h and out are L2-resident (12.8 MB each << 192 MB L2), so the ~205 MB
// gather + ~205 MB atomic traffic never touches HBM.
// ---------------------------------------------------------------------------
__global__ __launch_bounds__(256) void gnn_scatter_kernel(
    const float* __restrict__ h,          // [n_nodes, 64]
    const int*   __restrict__ edge_rows,  // [E]
    const int*   __restrict__ edge_cols,  // [E]
    const float* __restrict__ adj_vals,   // [E]
    float* __restrict__ out,              // [n_nodes, 64]
    int n_edges)
{
    int e = (int)(blockIdx.x * 8 + (threadIdx.x >> 5));
    e = __builtin_amdgcn_readfirstlane(e);           // wave-uniform -> SGPR
    if (e >= n_edges) return;

    const int lane = threadIdx.x & 31;
    const int r = edge_rows[e];                      // scalar (SMEM) loads
    const int c = edge_cols[e];
    const float a = adj_vals[e];

    const float2 hv = *(const float2*)(h + (size_t)c * DIM_OUT + lane * 2);
    float* op = out + (size_t)r * DIM_OUT + lane * 2;
    atomicAdd(op,     a * hv.x);                     // global_atomic_add_f32
    atomicAdd(op + 1, a * hv.y);
}

// ---------------------------------------------------------------------------
extern "C" void kernel_launch(void* const* d_in, const int* in_sizes, int n_in,
                              void* d_out, int out_size, void* d_ws, size_t ws_size,
                              hipStream_t stream)
{
    const float* x         = (const float*)d_in[0];   // [N,128] f32
    const float* W         = (const float*)d_in[1];   // [128,64] f32
    const int*   edge_rows = (const int*)  d_in[2];   // [E] i32
    const int*   edge_cols = (const int*)  d_in[3];   // [E] i32
    const float* adj_vals  = (const float*)d_in[4];   // [E] f32

    float* out = (float*)d_out;
    float* h   = (float*)d_ws;                        // n_nodes*64*4 = 12.8 MB scratch

    const int n_nodes = in_sizes[0] / DIM_IN;         // 50000
    const int n_edges = in_sizes[2];                  // 800000

    // Phase 1: GEMM. One 16x64 slab per wave; 8 waves per block.
    const int mtiles      = (n_nodes + 15) / 16;
    const int gemm_blocks = (mtiles + 7) / 8;
    gnn_gemm_kernel<<<gemm_blocks, 256, 0, stream>>>(x, W, h, n_nodes);

    // Phase 2: zero output (out_size = n_nodes*64, divisible by 4).
    const int n4 = out_size / 4;
    gnn_zero_kernel<<<(n4 + 255) / 256, 256, 0, stream>>>((float4*)d_out, n4);

    // Phase 3: scatter-add, one wave per edge, 8 waves per block.
    const int sc_blocks = (n_edges + 7) / 8;
    gnn_scatter_kernel<<<sc_blocks, 256, 0, stream>>>(
        h, edge_rows, edge_cols, adj_vals, out, n_edges);
}
